// MultiHeadAttention_7645041787133
// MI455X (gfx1250) — compile-verified, matrix operand order checked by reading, not run on hardware
//
#include <hip/hip_runtime.h>
#include <hip/hip_bf16.h>
#include <cstdint>

// MI455X / gfx1250, wave32. Matmuls via v_wmma_f32_16x16x32_f16; q tile staged
// into LDS by the Tensor Data Mover (tensor_load_to_lds + s_wait_tensorcnt).

typedef __attribute__((ext_vector_type(16))) _Float16 v16h;
typedef __attribute__((ext_vector_type(8)))  _Float16 v8h;
typedef __attribute__((ext_vector_type(8)))  float    v8f;
typedef __attribute__((ext_vector_type(4)))  float    v4f;
typedef __attribute__((ext_vector_type(4)))  unsigned int u32x4;
typedef __attribute__((ext_vector_type(8)))  int      i32x8;
typedef __attribute__((ext_vector_type(4)))  int      i32x4;

#define WMMA_F16(a, b, c) \
  __builtin_amdgcn_wmma_f32_16x16x32_f16(false, (a), false, (b), (short)0, (c), false, false)

#if __has_builtin(__builtin_amdgcn_tensor_load_to_lds) && \
    __has_builtin(__builtin_amdgcn_s_wait_tensorcnt)
#define USE_TDM 1
#else
#define USE_TDM 0
#endif

#define BATCH 8
#define SEQ   2048
#define EDIM  768          // concat dim
#define DDIM  64           // input dim
#define SCALING 0.125f     // 1/sqrt(64)

// LDS q-stage layout produced by TDM padding:
//   pad_interval = 128 dwords (256 halves), pad_amount = 4 dwords (8 halves)
//   -> chunk stride 264 halves, row stride 3*264 = 792 halves (bank-conflict-free)
#define Q_ROW   792
#define Q_CHUNK 264

// ---- WMMA fragment loaders -------------------------------------------------
// 16-bit A (16x32 MxK) / B (32x16 KxN) layout (ISA 7.12.2):
//   lane<16 : row/col = lane, K elements {0..7, 16..23}
//   lane>=16: row/col = lane-16, K elements {8..15, 24..31}

static __device__ __forceinline__ v16h frag_f16(const _Float16* base, int ld) {
  const int lane = threadIdx.x & 31;
  const int r = lane & 15;
  const int k = (lane & 16) ? 8 : 0;
  const _Float16* p = base + r * ld + k;
  v8h lo = *(const v8h*)(p);
  v8h hi = *(const v8h*)(p + 16);
  v16h f;
#pragma unroll
  for (int i = 0; i < 8; ++i) { f[i] = lo[i]; f[i + 8] = hi[i]; }
  return f;
}

static __device__ __forceinline__ v16h frag_f32(const float* base, int ld) {
  const int lane = threadIdx.x & 31;
  const int r = lane & 15;
  const int k = (lane & 16) ? 8 : 0;
  const float* p = base + r * ld + k;
  v4f x0 = *(const v4f*)(p);
  v4f x1 = *(const v4f*)(p + 4);
  v4f x2 = *(const v4f*)(p + 16);
  v4f x3 = *(const v4f*)(p + 20);
  v16h f;
#pragma unroll
  for (int i = 0; i < 4; ++i) {
    f[i]      = (_Float16)x0[i];
    f[i + 4]  = (_Float16)x1[i];
    f[i + 8]  = (_Float16)x2[i];
    f[i + 12] = (_Float16)x3[i];
  }
  return f;
}

// ---- Kernel 1: fused QKV projection, f32 -> f16, v stored transposed -------
__global__ __launch_bounds__(256) void qkv_kernel(
    const float* __restrict__ x,
    const float* __restrict__ w_q, const float* __restrict__ w_k,
    const float* __restrict__ w_v,
    _Float16* __restrict__ q16, _Float16* __restrict__ k16,
    _Float16* __restrict__ vT) {
  const int wave = threadIdx.x >> 5;
  const int lane = threadIdx.x & 31;
  const int g0 = blockIdx.x * 16;      // global row base (b*S + s)
  const int b  = g0 >> 11;
  const int s0 = g0 & (SEQ - 1);

  v16h a0 = frag_f32(x + (size_t)g0 * DDIM + 0,  DDIM);
  v16h a1 = frag_f32(x + (size_t)g0 * DDIM + 32, DDIM);

  const int M8 = (lane & 16) ? 8 : 0;
  const int N  = lane & 15;

  for (int t = 0; t < 18; ++t) {
    const int c = wave * 288 + t * 16;
    const float* w; int cl, which;
    if (c < 768)       { w = w_q; cl = c;        which = 0; }
    else if (c < 1536) { w = w_k; cl = c - 768;  which = 1; }
    else               { w = w_v; cl = c - 1536; which = 2; }

    v16h b0 = frag_f32(w + (size_t)cl * DDIM + 0,  DDIM);
    v16h b1 = frag_f32(w + (size_t)cl * DDIM + 32, DDIM);
    v8f acc = {};
    acc = WMMA_F16(a0, b0, acc);
    acc = WMMA_F16(a1, b1, acc);

    if (which == 0) {
#pragma unroll
      for (int i = 0; i < 8; ++i) acc[i] *= SCALING;   // fold 1/sqrt(d) into q
    }
    if (which == 2) {
#pragma unroll
      for (int i = 0; i < 8; ++i)
        vT[((size_t)b * EDIM + (cl + N)) * SEQ + (s0 + M8 + i)] = (_Float16)acc[i];
    } else {
      _Float16* dst = (which == 0) ? q16 : k16;
#pragma unroll
      for (int i = 0; i < 8; ++i)
        dst[(size_t)(g0 + M8 + i) * EDIM + cl + N] = (_Float16)acc[i];
    }
  }
}

// ---- Kernel 2: flash attention (online softmax) + final projection ---------
__global__ __launch_bounds__(256) void attn_kernel(
    const _Float16* __restrict__ q16, const _Float16* __restrict__ k16,
    const _Float16* __restrict__ vT,
    const float* __restrict__ w_final, const float* __restrict__ b_final,
    float* __restrict__ out) {
  __shared__ _Float16 qs[32 * Q_ROW];    // TDM-staged q tile [32 x 768] (padded)
  __shared__ float    S_s[32][68];       // score block [32 x 64]
  __shared__ _Float16 P_s[32][72];       // exp(S - m) in f16
  __shared__ _Float16 stage[32][832];    // O staged f16 (8 waves x 104-col windows)
  __shared__ float    outbuf[32][64];    // cross-wave f32 reduction
  __shared__ float    m_run[32], l_run[32], alpha_s[32], linv_s[32];

  const int tid  = threadIdx.x;
  const int wave = tid >> 5;
  const int lane = tid & 31;
  const int b    = blockIdx.x >> 6;
  const int qb   = blockIdx.x & 63;
  const int g0   = b * SEQ + qb * 32;    // global query-row base

  const int M8 = (lane & 16) ? 8 : 0;
  const int N  = lane & 15;

  if (tid < 32) { m_run[tid] = -3.0e38f; l_run[tid] = 0.0f; }
  for (int i = tid; i < 32 * 64; i += 256) ((float*)outbuf)[i] = 0.0f;

  // ---- stage q tile [32 x 768] f16 into LDS --------------------------------
#if USE_TDM
  if (wave == 0) {
    const unsigned long long ga =
        (unsigned long long)(uintptr_t)(q16 + (size_t)g0 * EDIM);
    const unsigned lds_off = (unsigned)(uintptr_t)(&qs[0]);
    u32x4 d0;
    d0[0] = 1u;                                   // count=1 valid descriptor
    d0[1] = lds_off;                              // lds_addr (bytes)
    d0[2] = (unsigned)(ga & 0xFFFFFFFFull);       // global_addr[31:0]
    d0[3] = (unsigned)((ga >> 32) & 0x1FFFFFFull) | (2u << 30);  // addr[56:32], type=2
    i32x8 d1;
    d1[0] = (int)((1u << 16)        // data_size = 2 bytes
            | (1u << 20)            // pad_enable
            | (6u << 22)            // pad_interval: 128 dwords
            | (3u << 25));          // pad_amount: 4 dwords
    d1[1] = (int)((EDIM & 0xFFFF) << 16);         // tensor_dim0 = 768 (lo16)
    d1[2] = (int)(((unsigned)(BATCH * SEQ) & 0xFFFF) << 16); // tensor_dim1 lo16
    d1[3] = (int)((((unsigned)(BATCH * SEQ)) >> 16) & 0xFFFF) | (EDIM << 16); // hi16 | tile_dim0
    d1[4] = 32;                                   // tile_dim1 = 32 rows
    d1[5] = EDIM;                                 // tensor_dim0_stride = 768
    d1[6] = 0;
    d1[7] = 0;
    i32x4 dz4 = {0, 0, 0, 0};
    i32x8 dz8 = {0, 0, 0, 0, 0, 0, 0, 0};
    __builtin_amdgcn_tensor_load_to_lds(d0, d1, dz4, dz4, dz8, 0);
    __builtin_amdgcn_s_wait_tensorcnt(0);
  }
#else
  for (int idx = tid; idx < 32 * 96; idx += 256) {
    const int row = idx / 96;
    const int col = (idx % 96) * 8;
    v8h val = *(const v8h*)(q16 + (size_t)(g0 + row) * EDIM + col);
    *(v8h*)(&qs[row * Q_ROW + (col / 256) * Q_CHUNK + (col % 256)]) = val;
  }
#endif
  __syncthreads();

  // score-tile assignment: wave -> (row half, key 16-col)
  const int s_tm = wave & 1;
  const int s_tn = wave >> 1;

  v8f O[2][6];
#pragma unroll
  for (int tm = 0; tm < 2; ++tm)
#pragma unroll
    for (int tc = 0; tc < 6; ++tc) O[tm][tc] = (v8f){};

  const _Float16* qbase = qs + (s_tm * 16) * Q_ROW;

  for (int j = 0; j < SEQ / 64; ++j) {
    const int k0 = j * 64;

    // ---- S = q . k^T  (this wave's 16x16 tile, K = 768) ----
    const _Float16* kb = k16 + (size_t)(b * SEQ + k0 + s_tn * 16) * EDIM;
    v8f sc = {};
#pragma unroll 4
    for (int kc = 0; kc < EDIM / 32; ++kc) {
      v16h a  = frag_f16(qbase + (kc >> 3) * Q_CHUNK + (kc & 7) * 32, Q_ROW);
      v16h bb = frag_f16(kb + kc * 32, EDIM);
      sc = WMMA_F16(a, bb, sc);
    }
#pragma unroll
    for (int i = 0; i < 8; ++i)
      S_s[s_tm * 16 + M8 + i][s_tn * 16 + N] = sc[i];

    if (j + 1 < SEQ / 64) {   // global_prefetch_b8 for next key block
      __builtin_prefetch(kb + 64 * EDIM, 0, 1);
      __builtin_prefetch(vT + ((size_t)b * EDIM + wave * 96) * SEQ + k0 + 64, 0, 1);
    }
    __syncthreads();

    // ---- online softmax over this 64-key block ----
    if (tid < 32) {
      float mo = m_run[tid], mx = mo;
      for (int n = 0; n < 64; ++n) mx = fmaxf(mx, S_s[tid][n]);
      float al = __expf(mo - mx);
      float sum = 0.0f;
      for (int n = 0; n < 64; ++n) {
        float p = __expf(S_s[tid][n] - mx);
        P_s[tid][n] = (_Float16)p;
        sum += p;
      }
      m_run[tid] = mx;
      l_run[tid] = l_run[tid] * al + sum;
      alpha_s[tid] = al;
    }
    __syncthreads();

    // ---- rescale O, accumulate O += P . V  (wave's 96-col slice) ----
    const int e0 = wave * 96;
    v16h pa[2][2];
#pragma unroll
    for (int tm = 0; tm < 2; ++tm)
#pragma unroll
      for (int kc = 0; kc < 2; ++kc)
        pa[tm][kc] = frag_f16(&P_s[tm * 16][0] + kc * 32, 72);

#pragma unroll
    for (int tm = 0; tm < 2; ++tm) {
#pragma unroll
      for (int tc = 0; tc < 6; ++tc) {
#pragma unroll
        for (int i = 0; i < 8; ++i)
          O[tm][tc][i] *= alpha_s[tm * 16 + M8 + i];
#pragma unroll
        for (int kc = 0; kc < 2; ++kc) {
          v16h vb = frag_f16(
              vT + ((size_t)b * EDIM + e0 + tc * 16) * SEQ + k0 + kc * 32, SEQ);
          O[tm][tc] = WMMA_F16(pa[tm][kc], vb, O[tm][tc]);
        }
      }
    }
    __syncthreads();   // protect S_s / P_s / alpha_s for next iteration
  }

  // ---- normalize and stage O as f16 ----
  if (tid < 32) linv_s[tid] = 1.0f / l_run[tid];
  __syncthreads();
#pragma unroll
  for (int tm = 0; tm < 2; ++tm)
#pragma unroll
    for (int tc = 0; tc < 6; ++tc)
#pragma unroll
      for (int i = 0; i < 8; ++i) {
        const int m = tm * 16 + M8 + i;
        stage[m][wave * 104 + tc * 16 + N] = (_Float16)(O[tm][tc][i] * linv_s[m]);
      }
  __syncthreads();

  // ---- final projection: out = O @ w_final^T + b  (WMMA, per-wave partials) ----
  v8f Fp[2][4];
#pragma unroll
  for (int tm = 0; tm < 2; ++tm)
#pragma unroll
    for (int td = 0; td < 4; ++td) Fp[tm][td] = (v8f){};

#pragma unroll
  for (int kc = 0; kc < 3; ++kc) {    // wave's 96-col K slice
    v16h a0 = frag_f16(&stage[0][0]  + wave * 104 + kc * 32, 832);
    v16h a1 = frag_f16(&stage[16][0] + wave * 104 + kc * 32, 832);
#pragma unroll
    for (int td = 0; td < 4; ++td) {
      v16h wb = frag_f32(w_final + (size_t)(td * 16) * EDIM + wave * 96 + kc * 32, EDIM);
      Fp[0][td] = WMMA_F16(a0, wb, Fp[0][td]);
      Fp[1][td] = WMMA_F16(a1, wb, Fp[1][td]);
    }
  }
#pragma unroll
  for (int tm = 0; tm < 2; ++tm)
#pragma unroll
    for (int td = 0; td < 4; ++td)
#pragma unroll
      for (int i = 0; i < 8; ++i)
        atomicAdd(&outbuf[tm * 16 + M8 + i][td * 16 + N], Fp[tm][td][i]);
  __syncthreads();

  for (int i = tid; i < 32 * 64; i += 256) {
    const int m = i >> 6, d = i & 63;
    out[(size_t)(g0 + m) * DDIM + d] = outbuf[m][d] + b_final[d];
  }
}

// ---- host glue -------------------------------------------------------------
extern "C" void kernel_launch(void* const* d_in, const int* in_sizes, int n_in,
                              void* d_out, int out_size, void* d_ws, size_t ws_size,
                              hipStream_t stream) {
  const float* x       = (const float*)d_in[0];
  const float* w_q     = (const float*)d_in[1];
  const float* w_k     = (const float*)d_in[2];
  const float* w_v     = (const float*)d_in[3];
  const float* w_final = (const float*)d_in[4];
  const float* b_final = (const float*)d_in[5];
  float* out = (float*)d_out;

  const size_t qkv_elems = (size_t)BATCH * SEQ * EDIM;
  _Float16* q16 = (_Float16*)d_ws;
  _Float16* k16 = q16 + qkv_elems;
  _Float16* vT  = k16 + qkv_elems;

  qkv_kernel<<<(BATCH * SEQ) / 16, 256, 0, stream>>>(x, w_q, w_k, w_v, q16, k16, vT);
  attn_kernel<<<BATCH * (SEQ / 32), 256, 0, stream>>>(q16, k16, vT, w_final, b_final, out);
}
